// ApicalPathway_14448269984072
// MI455X (gfx1250) — matrix operand order repaired and checked by reading, not compiled, not run on hardware
//
#include <hip/hip_runtime.h>
#include <hip/hip_bf16.h>

// Problem dims (fixed by the reference)
#define BATCH 32
#define NCOL  1024
#define EDIM  128
#define ODIM  128

typedef __attribute__((ext_vector_type(16))) __bf16 v16bf;
typedef __attribute__((ext_vector_type(8)))  float  v8f;
typedef __attribute__((ext_vector_type(4)))  unsigned int v4u;

union Frag32B {           // 32 bytes: one WMMA A or B operand (16 x bf16)
    v16bf bf;
    v4u   q[2];
};

__device__ __forceinline__ unsigned short f32_to_bf16_rne(float f) {
    unsigned int u = __float_as_uint(f);
    u += 0x7FFFu + ((u >> 16) & 1u);     // round-to-nearest-even
    return (unsigned short)(u >> 16);
}

// gate(x) = 1 + tanh(x) = 2 - 2/(exp(2x)+1); safe at +/-inf.
__device__ __forceinline__ float gate_1p_tanh(float x) {
    const float LOG2E_X2 = 2.8853900817779268f;   // 2*log2(e)
    float t = __builtin_amdgcn_exp2f(x * LOG2E_X2);   // exp(2x) via v_exp_f32
    return 2.0f - 2.0f * __builtin_amdgcn_rcpf(t + 1.0f);
}

__global__ __launch_bounds__(256) void apical_wmma_kernel(
    const float* __restrict__ I_l5e,     // [B, C, O]
    const float* __restrict__ thal,      // [B, C, E]
    const float* __restrict__ proj,      // [C, O, E]
    float* __restrict__ out)             // [B, C, O]
{
    const int c   = blockIdx.x;          // one column per workgroup
    const int tid = threadIdx.x;

    __shared__ __align__(16) unsigned short sA[BATCH * EDIM];  // 8 KB  bf16 [b][e]
    __shared__ __align__(16) unsigned short sB[ODIM * EDIM];   // 32 KB bf16 [o][e]

    // ---- Stage A = thal[:, c, :] -> LDS bf16 (coalesced b128 loads) ----
#pragma unroll
    for (int i = 0; i < 4; ++i) {
        int idx = tid + i * 256;                 // float4 index in 32x(128/4)
        int b   = idx >> 5;
        int e   = (idx & 31) << 2;
        const float4 f = *(const float4*)(thal + ((size_t)b * NCOL + c) * EDIM + e);
        uint2 p;
        p.x = (unsigned)f32_to_bf16_rne(f.x) | ((unsigned)f32_to_bf16_rne(f.y) << 16);
        p.y = (unsigned)f32_to_bf16_rne(f.z) | ((unsigned)f32_to_bf16_rne(f.w) << 16);
        *(uint2*)(&sA[b * EDIM + e]) = p;
    }
    // ---- Stage B = proj[c] (contiguous 64 KB block) -> LDS bf16 ----
    const float* projc = proj + (size_t)c * ODIM * EDIM;
#pragma unroll
    for (int i = 0; i < 16; ++i) {
        int idx = tid + i * 256;                 // float4 index in 128x(128/4)
        int o   = idx >> 5;
        int e   = (idx & 31) << 2;
        const float4 f = *(const float4*)(projc + o * EDIM + e);
        uint2 p;
        p.x = (unsigned)f32_to_bf16_rne(f.x) | ((unsigned)f32_to_bf16_rne(f.y) << 16);
        p.y = (unsigned)f32_to_bf16_rne(f.z) | ((unsigned)f32_to_bf16_rne(f.w) << 16);
        *(uint2*)(&sB[o * EDIM + e]) = p;
    }
    __syncthreads();

    // ---- Tile assignment: 2 M-tiles x 8 N-tiles; wave w -> M=w&1, N pair = w>>1 ----
    const int lane  = tid & 31;
    const int wave  = tid >> 5;
    const int mBase = (wave & 1) * 16;           // 0 or 16 (batch rows)
    const int n0    = (wave >> 1) * 32;          // two adjacent 16-wide N tiles
    const int half  = lane >> 4;                 // 0: lanes 0-15, 1: lanes 16-31
    const int lsub  = lane & 15;

    const int aRow = mBase + lsub;               // A-matrix row M (ISA layout: M = lane%16)
    const int nRow0 = n0 + lsub;                 // B-matrix column N for tile 0
    const int nRow1 = n0 + 16 + lsub;            // ... for tile 1

    v8f acc0 = {};
    v8f acc1 = {};

#pragma unroll
    for (int ks = 0; ks < 4; ++ks) {             // K = 128 in steps of 32
        const int kb = ks * 32;
        Frag32B a, b0, b1;
        // A 16-bit 16x32 layout: lane group holds K = half*16 + 0..15 (contiguous in e), matching B
        a.q[0]  = *(const v4u*)(&sA[aRow * EDIM + kb + half * 16]);
        a.q[1]  = *(const v4u*)(&sA[aRow * EDIM + kb + half * 16 + 8]);
        // B 32x16 layout: lane group holds K = half*16 + 0..15 (contiguous in e) for its N
        b0.q[0] = *(const v4u*)(&sB[nRow0 * EDIM + kb + half * 16]);
        b0.q[1] = *(const v4u*)(&sB[nRow0 * EDIM + kb + half * 16 + 8]);
        b1.q[0] = *(const v4u*)(&sB[nRow1 * EDIM + kb + half * 16]);
        b1.q[1] = *(const v4u*)(&sB[nRow1 * EDIM + kb + half * 16 + 8]);

        acc0 = __builtin_amdgcn_wmma_f32_16x16x32_bf16(
                   false, a.bf, false, b0.bf, (short)0, acc0, false, false);
        acc1 = __builtin_amdgcn_wmma_f32_16x16x32_bf16(
                   false, a.bf, false, b1.bf, (short)0, acc1, false, false);
    }

    // ---- Epilogue: D layout VGPR r -> M = mBase + half*8 + r, N = tile + lsub ----
    const int bRow  = mBase + half * 8;
    const int oCol0 = n0 + lsub;
    const int oCol1 = n0 + 16 + lsub;
#pragma unroll
    for (int r = 0; r < 8; ++r) {
        const int b = bRow + r;
        const size_t base = ((size_t)b * NCOL + c) * ODIM;
        out[base + oCol0] = I_l5e[base + oCol0] * gate_1p_tanh(acc0[r]);
        out[base + oCol1] = I_l5e[base + oCol1] * gate_1p_tanh(acc1[r]);
    }
}

extern "C" void kernel_launch(void* const* d_in, const int* in_sizes, int n_in,
                              void* d_out, int out_size, void* d_ws, size_t ws_size,
                              hipStream_t stream) {
    const float* I_l5e = (const float*)d_in[0];   // [32,1024,128]
    const float* thal  = (const float*)d_in[1];   // [32,1024,128]
    const float* proj  = (const float*)d_in[2];   // [1024,128,128]
    float* out         = (float*)d_out;           // [32,1024,128]

    apical_wmma_kernel<<<NCOL, 256, 0, stream>>>(I_l5e, thal, proj, out);
}
